// MambaLayer_31258771980442
// MI455X (gfx1250) — matrix operand order repaired and checked by reading, not compiled, not run on hardware
//
#include <hip/hip_runtime.h>
#include <hip/hip_bf16.h>

// ---------------------------------------------------------------------------
// Mamba layer forward for gfx1250 (MI455X), bf16 WMMA data path.
// All GEMM operands pre-converted to bf16 (A row-major, B transposed N-major)
// so every WMMA fragment is exactly two global_load_b128s — no scalar gathers
// or float->bf16 conversion in the hot loops.
// ---------------------------------------------------------------------------

#define BSZ      2
#define LSEQ     2048
#define DMODEL   1024
#define DINNER   2048
#define DSTATE   8
#define DCONV    4
#define DTRANK   64
#define MROWS    (BSZ * LSEQ)          // 4096
#define XDBL_LD  (DTRANK + 2 * DSTATE) // 80

typedef __attribute__((ext_vector_type(16))) __bf16 v16bf;
typedef __attribute__((ext_vector_type(8)))  __bf16 v8bf;
typedef __attribute__((ext_vector_type(8)))  float  v8f;

// ---- WMMA fragment loaders (layouts per CDNA5 ISA 7.12.2, wave32) ----------
// A: 16x32 bf16 row-major. lane holds row M = lane%16;
// element e -> K = e + 8*((e>>3) + (lane>>4))  => two contiguous 8-elem runs.
__device__ __forceinline__ v16bf load_A16(const __bf16* __restrict__ A, int lda,
                                          int m0, int k0, int lane) {
    const __bf16* p = A + (long)(m0 + (lane & 15)) * lda + k0 + 8 * (lane >> 4);
    const v8bf lo = *(const v8bf*)p;         // K = k0+8hi   .. +8
    const v8bf hi = *(const v8bf*)(p + 16);  // K = k0+8hi+16 .. +8
    return __builtin_shufflevector(lo, hi, 0, 1, 2, 3, 4, 5, 6, 7,
                                           8, 9, 10, 11, 12, 13, 14, 15);
}

// B: 32x16 bf16 from transposed (N-major) weights Bt[N][K].
// lane holds col N = lane%16; element e -> K = e + 8*((e>>3) + (lane>>4)) (same as A).
__device__ __forceinline__ v16bf load_Bt16(const __bf16* __restrict__ Bt, int ldbt,
                                           int n0, int k0, int lane) {
    const __bf16* p = Bt + (long)(n0 + (lane & 15)) * ldbt + k0 + 8 * (lane >> 4);
    const v8bf lo = *(const v8bf*)p;
    const v8bf hi = *(const v8bf*)(p + 16);
    return __builtin_shufflevector(lo, hi, 0, 1, 2, 3, 4, 5, 6, 7,
                                           8, 9, 10, 11, 12, 13, 14, 15);
}

// ---- WMMA GEMM: C = epi(A[M,K] @ Bt[N,K]^T + bias) -------------------------
// One wave computes a (16*MT) x (16*NT) tile. EPI: 0 none, 1 bias+silu,
// 2 bias+softplus. WF writes f32 C, WH writes bf16 C (either or both).
template <int MT, int NT, int EPI, bool WF, bool WH>
__global__ __launch_bounds__(32)
void wmma_gemm16(const __bf16* __restrict__ A, const __bf16* __restrict__ Bt,
                 const float* __restrict__ bias,
                 float* __restrict__ Cf, __bf16* __restrict__ Ch,
                 int K, int lda, int ldbt, int ldc) {
    const int lane = threadIdx.x & 31;
    const int n0 = blockIdx.x * (16 * NT);
    const int m0 = blockIdx.y * (16 * MT);

    v8f acc[MT][NT];
#pragma unroll
    for (int m = 0; m < MT; ++m)
#pragma unroll
        for (int t = 0; t < NT; ++t)
#pragma unroll
            for (int v = 0; v < 8; ++v) acc[m][t][v] = 0.0f;

    for (int k0 = 0; k0 < K; k0 += 32) {
        v16bf a[MT], b[NT];
#pragma unroll
        for (int m = 0; m < MT; ++m) a[m] = load_A16(A, lda, m0 + 16 * m, k0, lane);
#pragma unroll
        for (int t = 0; t < NT; ++t) b[t] = load_Bt16(Bt, ldbt, n0 + 16 * t, k0, lane);
#pragma unroll
        for (int m = 0; m < MT; ++m)
#pragma unroll
            for (int t = 0; t < NT; ++t)
                acc[m][t] = __builtin_amdgcn_wmma_f32_16x16x32_bf16(
                    false, a[m], false, b[t], (short)0, acc[m][t], false, false);
    }

    // C/D layout: VGPR v -> row 16*m + v + 8*(lane>>4), col 16*t + lane%16
    const int hi = lane >> 4;
    const int cb = lane & 15;
#pragma unroll
    for (int m = 0; m < MT; ++m)
#pragma unroll
        for (int t = 0; t < NT; ++t) {
            const int n = n0 + 16 * t + cb;
            float bv = 0.0f;
            if (EPI != 0) bv = bias[n];
#pragma unroll
            for (int v = 0; v < 8; ++v) {
                const long row = m0 + 16 * m + v + 8 * hi;
                float x = acc[m][t][v];
                if (EPI != 0) x += bv;                                  // bias
                if (EPI == 1) x = x / (1.0f + __expf(-x));              // silu
                if (EPI == 2) x = (x > 20.0f) ? x : log1pf(__expf(x));  // softplus
                if (WF) Cf[row * ldc + n] = x;
                if (WH) Ch[row * ldc + n] = (__bf16)x;
            }
        }
}

// ---- One-time operand conversions ------------------------------------------
__global__ void f32_to_bf16_kernel(const float* __restrict__ src,
                                   __bf16* __restrict__ dst, long n) {
    const long i = (long)blockIdx.x * blockDim.x + threadIdx.x;
    if (i < n) dst[i] = (__bf16)src[i];
}

// W[K][N] f32 -> Wt[N][K] bf16
__global__ void transpose_to_bf16_kernel(const float* __restrict__ W,
                                         __bf16* __restrict__ Wt, int K, int N) {
    const long i = (long)blockIdx.x * blockDim.x + threadIdx.x;
    if (i >= (long)K * N) return;
    const int n = (int)(i % N);
    const int k = (int)(i / N);
    Wt[(long)n * K + k] = (__bf16)W[i];
}

// ---- Causal depthwise conv (k=4) over L ------------------------------------
// Reads x_in = first DINNER columns of bf16 xz (row stride 2*DINNER).
__global__ void dwconv_kernel(const __bf16* __restrict__ xz16,
                              const float* __restrict__ dw,    // [DCONV][DINNER]
                              __bf16* __restrict__ xconv16) {
    const long idx = (long)blockIdx.x * blockDim.x + threadIdx.x;
    if (idx >= (long)MROWS * DINNER) return;
    const int  c  = (int)(idx % DINNER);
    const long bl = idx / DINNER;
    const int  l  = (int)(bl % LSEQ);
    const long b0 = bl - l;                    // b * LSEQ
    float acc = 0.0f;
#pragma unroll
    for (int j = 0; j < DCONV; ++j) {
        const int t = l - (DCONV - 1) + j;
        if (t >= 0)
            acc += (float)xz16[(b0 + t) * (2L * DINNER) + c] * dw[j * DINNER + c];
    }
    xconv16[idx] = (__bf16)acc;
}

// ---- Selective scan --------------------------------------------------------
// 8 consecutive lanes carry the 8 SSM states of one (b,d) channel; y_t is
// reduced with 3 xor-shuffles (wave32). State kept in f32 registers.
__global__ void ssm_scan(const __bf16* __restrict__ delta16,  // [MROWS][DINNER]
                         const float*  __restrict__ xdbl,     // [MROWS][80]
                         const __bf16* __restrict__ xc16,     // [MROWS][DINNER]
                         const float*  __restrict__ A_log,    // [DINNER][DSTATE]
                         const float*  __restrict__ Dp,       // [DINNER]
                         const __bf16* __restrict__ xz16,     // z = cols [DINNER,2*DINNER)
                         __bf16* __restrict__ y16) {          // [MROWS][DINNER]
    const int gid = blockIdx.x * blockDim.x + threadIdx.x;
    const int n   = gid & (DSTATE - 1);
    const int bd  = gid >> 3;
    if (bd >= BSZ * DINNER) return;
    const int d = bd & (DINNER - 1);
    const int b = bd >> 11;                    // / DINNER

    const float An = -__expf(A_log[d * DSTATE + n]);
    const float Dd = Dp[d];
    float h = 0.0f;

    const long row0 = (long)b * LSEQ;
    for (int t = 0; t < LSEQ; ++t) {
        const long r   = row0 + t;
        const float dt  = (float)delta16[r * DINNER + d];
        const float xcv = (float)xc16[r * DINNER + d];
        const float Bn  = xdbl[r * XDBL_LD + DTRANK + n];
        const float Cn  = xdbl[r * XDBL_LD + DTRANK + DSTATE + n];

        h = __expf(dt * An) * h + dt * Bn * xcv;
        float y = Cn * h;
        y += __shfl_xor(y, 1, 32);
        y += __shfl_xor(y, 2, 32);
        y += __shfl_xor(y, 4, 32);
        if (n == 0) {
            float yv = y + xcv * Dd;
            const float z = (float)xz16[r * (2L * DINNER) + DINNER + d];
            yv *= z / (1.0f + __expf(-z));     // * silu(z)
            y16[r * DINNER + d] = (__bf16)yv;
        }
    }
}

// ---------------------------------------------------------------------------
extern "C" void kernel_launch(void* const* d_in, const int* in_sizes, int n_in,
                              void* d_out, int out_size, void* d_ws, size_t ws_size,
                              hipStream_t stream) {
    const float* x         = (const float*)d_in[0];   // [B,L,DMODEL]
    const float* W_in      = (const float*)d_in[1];   // [DMODEL, 2*DINNER]
    const float* dw_kernel = (const float*)d_in[2];   // [DCONV, DINNER]
    const float* pw_kernel = (const float*)d_in[3];   // [DINNER, DINNER]
    const float* conv_bias = (const float*)d_in[4];   // [DINNER]
    const float* W_x       = (const float*)d_in[5];   // [DINNER, 80]
    const float* W_dt      = (const float*)d_in[6];   // [DTRANK, DINNER]
    const float* b_dt      = (const float*)d_in[7];   // [DINNER]
    const float* A_log     = (const float*)d_in[8];   // [DINNER, DSTATE]
    const float* Dvec      = (const float*)d_in[9];   // [DINNER]
    const float* W_out     = (const float*)d_in[10];  // [DINNER, DMODEL]
    float* out = (float*)d_out;                       // [B,L,DMODEL] f32

    // ---- workspace layout (~127 MB) ----
    char* p = (char*)d_ws;
    __bf16* x16     = (__bf16*)p; p += (long)MROWS * DMODEL * 2;       //  8 MB
    __bf16* WinT    = (__bf16*)p; p += (long)DMODEL * 2 * DINNER * 2;  //  8 MB
    __bf16* pwT     = (__bf16*)p; p += (long)DINNER * DINNER * 2;      //  8 MB
    __bf16* WxT     = (__bf16*)p; p += (long)XDBL_LD * DINNER * 2;     //  0.3 MB
    __bf16* WdtT    = (__bf16*)p; p += (long)DINNER * DTRANK * 2;      //  0.25 MB
    __bf16* WoutT   = (__bf16*)p; p += (long)DMODEL * DINNER * 2;      //  4 MB
    __bf16* xz16    = (__bf16*)p; p += (long)MROWS * 2 * DINNER * 2;   // 32 MB
    __bf16* xconv16 = (__bf16*)p; p += (long)MROWS * DINNER * 2;       // 16 MB
    __bf16* xc16    = (__bf16*)p; p += (long)MROWS * DINNER * 2;       // 16 MB
    float*  xdbl    = (float*) p; p += (long)MROWS * XDBL_LD * 4;      //  1.25 MB
    __bf16* xdbl16  = (__bf16*)p; p += (long)MROWS * XDBL_LD * 2;      //  0.6 MB
    __bf16* delta16 = (__bf16*)p; p += (long)MROWS * DINNER * 2;       // 16 MB
    __bf16* y16     = (__bf16*)p; p += (long)MROWS * DINNER * 2;       // 16 MB

    const dim3 wave(32);
    auto blocks1d = [](long n) { return (unsigned)((n + 255) / 256); };

    // ---- 0) operand conversions (bf16, weights transposed to N-major) ----
    f32_to_bf16_kernel<<<blocks1d((long)MROWS * DMODEL), 256, 0, stream>>>(
        x, x16, (long)MROWS * DMODEL);
    transpose_to_bf16_kernel<<<blocks1d((long)DMODEL * 2 * DINNER), 256, 0, stream>>>(
        W_in, WinT, DMODEL, 2 * DINNER);
    transpose_to_bf16_kernel<<<blocks1d((long)DINNER * DINNER), 256, 0, stream>>>(
        pw_kernel, pwT, DINNER, DINNER);
    transpose_to_bf16_kernel<<<blocks1d((long)DINNER * XDBL_LD), 256, 0, stream>>>(
        W_x, WxT, DINNER, XDBL_LD);
    transpose_to_bf16_kernel<<<blocks1d((long)DTRANK * DINNER), 256, 0, stream>>>(
        W_dt, WdtT, DTRANK, DINNER);
    transpose_to_bf16_kernel<<<blocks1d((long)DINNER * DMODEL), 256, 0, stream>>>(
        W_out, WoutT, DINNER, DMODEL);

    // ---- 1) xz = x @ W_in                       [4096 x 4096 x 1024] ----
    wmma_gemm16<2, 4, 0, false, true>
        <<<dim3((2 * DINNER) / 64, MROWS / 32), wave, 0, stream>>>(
        x16, WinT, nullptr, nullptr, xz16, DMODEL, DMODEL, DMODEL, 2 * DINNER);

    // ---- 2) depthwise causal conv ----
    dwconv_kernel<<<blocks1d((long)MROWS * DINNER), 256, 0, stream>>>(
        xz16, dw_kernel, xconv16);

    // ---- 3) xc = silu(xconv @ pw + bias)        [4096 x 2048 x 2048] ----
    wmma_gemm16<2, 4, 1, false, true>
        <<<dim3(DINNER / 64, MROWS / 32), wave, 0, stream>>>(
        xconv16, pwT, conv_bias, nullptr, xc16, DINNER, DINNER, DINNER, DINNER);

    // ---- 4) x_dbl = xc @ W_x                    [4096 x 80 x 2048] ----
    wmma_gemm16<2, 1, 0, true, true>
        <<<dim3(XDBL_LD / 16, MROWS / 32), wave, 0, stream>>>(
        xc16, WxT, nullptr, xdbl, xdbl16, DINNER, DINNER, DINNER, XDBL_LD);

    // ---- 5) delta = softplus(x_dbl[:,:64] @ W_dt + b_dt)  [4096x2048x64] ----
    wmma_gemm16<2, 4, 2, false, true>
        <<<dim3(DINNER / 64, MROWS / 32), wave, 0, stream>>>(
        xdbl16, WdtT, b_dt, nullptr, delta16, DTRANK, XDBL_LD, DTRANK, DINNER);

    // ---- 6) selective scan (+ xc*D, * silu(z)) -> y16 ----
    ssm_scan<<<(BSZ * DINNER * DSTATE) / 256, 256, 0, stream>>>(
        delta16, xdbl, xc16, A_log, Dvec, xz16, y16);

    // ---- 7) out = y @ W_out                     [4096 x 1024 x 2048] ----
    wmma_gemm16<2, 4, 0, true, false>
        <<<dim3(DMODEL / 64, MROWS / 32), wave, 0, stream>>>(
        y16, WoutT, nullptr, out, nullptr, DINNER, DINNER, DINNER, DMODEL);
}
